// ST_HMR_78082505441385
// MI455X (gfx1250) — matrix operand order repaired and checked by reading, not compiled, not run on hardware
//
#include <hip/hip_runtime.h>
#include <hip/hip_bf16.h>
#include <math.h>

// ---------------------------------------------------------------------------
// Types for CDNA5 WMMA (gfx1250, wave32)
// ---------------------------------------------------------------------------
typedef __attribute__((ext_vector_type(16))) __bf16 bf16x16;
typedef __attribute__((ext_vector_type(8)))  __bf16 bf16x8;
typedef __attribute__((ext_vector_type(8)))  float  v8f;

// ---------------------------------------------------------------------------
// Problem constants
// ---------------------------------------------------------------------------
constexpr int Bb = 16, Tt = 49, Nj = 24, Hh = 256;
constexpr int Mrows = Bb * Tt;              // 784 = 49 tiles of 16
constexpr int K1 = 7 * Hh;                  // 1792 (p | h_tb | h | h_ta | h_sb | g_t | g_s)
constexpr int K2 = 2 * Hh;                  // 512  (h_new | g)
constexpr long long E = (long long)Bb * Tt * Nj * Hh;   // 4,816,896 elems per tensor

// workspace layout (element counts)
constexpr long long SZ_WB   = 9LL * Nj * Hh * K1;         // bf16 weights pass1, col-major per (g,n)
constexpr long long SZ_WB2  = 6LL * Nj * Hh * K2;         // bf16 weights pass2
constexpr long long SZ_XB   = (long long)Nj * Mrows * K1; // bf16 X pass1
constexpr long long SZ_XB2  = 2LL * Nj * Mrows * K2;      // bf16 X pass2 (t-path, s-path)
constexpr size_t OFF_WB   = 0;
constexpr size_t OFF_WB2  = OFF_WB   + (size_t)SZ_WB  * 2;
constexpr size_t OFF_XB   = OFF_WB2  + (size_t)SZ_WB2 * 2;
constexpr size_t OFF_XB2  = OFF_XB   + (size_t)SZ_XB  * 2;

static __device__ __forceinline__ float sigf(float x) { return 1.0f / (1.0f + __expf(-x)); }

// ---------------------------------------------------------------------------
// Prep: weights pass 1 -> bf16, column-major [g*24+n][col][k], k-segments:
//   0..255 U | 256..1023 Wt(3H) | 1024..1279 Ws | 1280..1535 Zt | 1536..1791 Zs_eff
// Faithful quirk: gate 2 (f_ft) uses Zs[1].
// ---------------------------------------------------------------------------
__global__ void __launch_bounds__(256) prep_w1(
    const float* __restrict__ U,  const float* __restrict__ Wt,
    const float* __restrict__ Ws, const float* __restrict__ Zt,
    const float* __restrict__ Zs, __bf16* __restrict__ Wb) {
  long long idx = (long long)blockIdx.x * blockDim.x + threadIdx.x;
  if (idx >= SZ_WB) return;
  int k = (int)(idx % K1);
  int c = (int)((idx / K1) % Hh);
  int n = (int)((idx / ((long long)K1 * Hh)) % Nj);
  int g = (int)(idx / ((long long)K1 * Hh * Nj));
  float v;
  size_t gn = (size_t)g * Nj + n;
  if (k < 256)        v = U [(gn * 256 + k)          * 256 + c];
  else if (k < 1024)  v = Wt[(gn * 768 + (k - 256))  * 256 + c];
  else if (k < 1280)  v = Ws[(gn * 256 + (k - 1024)) * 256 + c];
  else if (k < 1536)  v = Zt[(gn * 256 + (k - 1280)) * 256 + c];
  else {
    int gz = (g == 2) ? 1 : g;
    v = Zs[(((size_t)gz * Nj + n) * 256 + (k - 1536)) * 256 + c];
  }
  Wb[idx] = (__bf16)v;
}

// Prep: weights pass 2 -> bf16 col-major, k: 0..255 Wg | 256..511 Zg
__global__ void __launch_bounds__(256) prep_w2(
    const float* __restrict__ Wg, const float* __restrict__ Zg,
    __bf16* __restrict__ Wb2) {
  long long idx = (long long)blockIdx.x * blockDim.x + threadIdx.x;
  if (idx >= SZ_WB2) return;
  int k = (int)(idx % K2);
  int c = (int)((idx / K2) % Hh);
  int n = (int)((idx / ((long long)K2 * Hh)) % Nj);
  int g = (int)(idx / ((long long)K2 * Hh * Nj));
  size_t gn = (size_t)g * Nj + n;
  float v = (k < 256) ? Wg[(gn * 256 + k)         * 256 + c]
                      : Zg[(gn * 256 + (k - 256)) * 256 + c];
  Wb2[idx] = (__bf16)v;
}

// Prep: X pass 1 -> bf16 [n][row=b*T+t][K1] with spatial/temporal shifts
__global__ void __launch_bounds__(256) prep_x1(
    const float* __restrict__ p, const float* __restrict__ h,
    const float* __restrict__ g_t, const float* __restrict__ g_s,
    __bf16* __restrict__ Xb) {
  long long idx = (long long)blockIdx.x * blockDim.x + threadIdx.x;
  if (idx >= SZ_XB) return;
  int k   = (int)(idx % K1);
  int row = (int)((idx / K1) % Mrows);
  int n   = (int)(idx / ((long long)K1 * Mrows));
  int t   = row % Tt;
  size_t base = ((size_t)row * Nj + n) * Hh;
  float v;
  if (k < 256)        v = p[base + k];
  else if (k < 512)   v = (n > 0)      ? h[base - 256 + (k - 256)]        : 0.0f;
  else if (k < 768)   v = h[base + (k - 512)];
  else if (k < 1024)  v = (n < Nj - 1) ? h[base + 256 + (k - 768)]        : 0.0f;
  else if (k < 1280)  v = (t > 0)      ? h[base - (size_t)Nj * Hh + (k - 1024)] : 0.0f;
  else if (k < 1536)  v = g_t[base + (k - 1280)];
  else                v = g_s[base + (k - 1536)];
  Xb[idx] = (__bf16)v;
}

// Prep: pass-2 X g-halves: X2[path*24+n][row][256 + h] = bf16(g_{t|s})
// (h_new halves are written by the fused pass-1 epilogue)
__global__ void __launch_bounds__(256) prep_x2g(
    const float* __restrict__ g_t, const float* __restrict__ g_s,
    __bf16* __restrict__ X2) {
  long long idx = (long long)blockIdx.x * blockDim.x + threadIdx.x;
  if (idx >= 2 * E) return;
  int path = (int)(idx / E);
  long long e = idx - (long long)path * E;
  int hcol = (int)(e & 255);
  int n    = (int)((e >> 8) % Nj);
  int row  = (int)(e / ((long long)Nj * Hh));
  const float* src = path ? g_s : g_t;
  X2[(((size_t)path * Nj + n) * Mrows + row) * K2 + 256 + hcol] = (__bf16)src[e];
}

// ---------------------------------------------------------------------------
// Fused pass 1: one wave = 16x16 output tile of joint n, ALL 9 gates
// (9 accumulators sharing one A fragment), then cell update in the epilogue.
// Gate preactivations never touch memory.
// A/B/C fragment layouts per CDNA5 ISA 16-bit WMMA tables.
// ---------------------------------------------------------------------------
__global__ void __launch_bounds__(256) fused_gate_gemm(
    const __bf16* __restrict__ X, const __bf16* __restrict__ W,
    const float* __restrict__ bias,
    const float* __restrict__ c_h, const float* __restrict__ c_g_t,
    const float* __restrict__ c_g_s,
    float* __restrict__ out, __bf16* __restrict__ X2) {
  int wave = blockIdx.x * (blockDim.x >> 5) + (threadIdx.x >> 5);
  // waves = 24 * 49 * 16 = 18816 (grid exact, no guard needed; keep one anyway)
  if (wave >= Nj * 49 * 16) return;
  int n   = wave / (49 * 16);
  int rem = wave - n * (49 * 16);
  int mt = rem >> 4;          // 0..48 row tile
  int ct = rem & 15;          // 0..15 col tile (16 cols)
  int lane = threadIdx.x & 31;
  int half = lane >> 4;
  int l    = lane & 15;
  int col  = ct * 16 + l;

  const __bf16* Xrow = X + ((size_t)n * Mrows + (size_t)(mt * 16 + l)) * K1;

  v8f acc[9] = {};
  for (int k0 = 0; k0 < K1; k0 += 32) {
    const __bf16* ap = Xrow + k0 + half * 8;
    bf16x8 alo = *(const bf16x8*)(ap);
    bf16x8 ahi = *(const bf16x8*)(ap + 16);
    bf16x16 a = __builtin_shufflevector(alo, ahi,
        0,1,2,3,4,5,6,7,8,9,10,11,12,13,14,15);
#pragma unroll
    for (int g = 0; g < 9; ++g) {
      const __bf16* bp = W + (((size_t)g * Nj + n) * Hh + col) * K1 + k0 + half * 8;
      bf16x8 blo = *(const bf16x8*)(bp);
      bf16x8 bhi = *(const bf16x8*)(bp + 16);
      bf16x16 bfrag = __builtin_shufflevector(blo, bhi,
          0,1,2,3,4,5,6,7,8,9,10,11,12,13,14,15);
      acc[g] = __builtin_amdgcn_wmma_f32_16x16x32_bf16(
          false, a, false, bfrag, (short)0, acc[g], false, false);
    }
  }

  // ---- fused cell epilogue ----
  float bb[9];
#pragma unroll
  for (int g = 0; g < 9; ++g)
    bb[g] = bias[((size_t)g * Nj + n) * Hh + col];

#pragma unroll
  for (int r = 0; r < 8; ++r) {
    int row = mt * 16 + half * 8 + r;
    int t   = row % Tt;
    long long idx = ((long long)row * Nj + n) * Hh + col;

    float i_n  = sigf (acc[0][r] + bb[0]);
    float f_lt = sigf (acc[1][r] + bb[1]);
    float f_ft = sigf (acc[2][r] + bb[2]);
    float f_rt = sigf (acc[3][r] + bb[3]);
    float f_s  = sigf (acc[4][r] + bb[4]);
    float f_gt = sigf (acc[5][r] + bb[5]);
    float f_gs = sigf (acc[6][r] + bb[6]);
    float o_n  = sigf (acc[7][r] + bb[7]);
    float c_n  = tanhf(acc[8][r] + bb[8]);

    float ctb = (n > 0)      ? c_h[idx - Hh] : 0.0f;
    float cta = (n < Nj - 1) ? c_h[idx + Hh] : 0.0f;
    float csb = (t > 0)      ? c_h[idx - (long long)Nj * Hh] : 0.0f;

    float cnew = f_lt * ctb + f_ft * c_h[idx] + f_rt * cta + f_s * csb
               + f_gt * c_g_t[idx] + f_gs * c_g_s[idx] + i_n * c_n;
    float hnew = o_n * tanhf(cnew);

    out[idx]     = hnew;   // h_new
    out[E + idx] = cnew;   // c_h_new

    __bf16 hb = (__bf16)hnew;
    X2[((size_t)n * Mrows + row) * K2 + col]                 = hb;  // t-path h half
    X2[(((size_t)Nj + n) * Mrows + row) * K2 + col]          = hb;  // s-path h half
  }
}

// ---------------------------------------------------------------------------
// Fused pass 2: one wave = 16x16 tile of (path, n), 3 global gates
// (f, cand, o) sharing one A fragment; global-cell update in epilogue.
// ---------------------------------------------------------------------------
__global__ void __launch_bounds__(256) fused_global_gemm(
    const __bf16* __restrict__ X2, const __bf16* __restrict__ W2,
    const float* __restrict__ bg,
    const float* __restrict__ c_g_t, const float* __restrict__ c_g_s,
    float* __restrict__ out) {
  int wave = blockIdx.x * (blockDim.x >> 5) + (threadIdx.x >> 5);
  // waves = 2 * 24 * 49 * 16 = 37632
  if (wave >= 2 * Nj * 49 * 16) return;
  int path = wave / (Nj * 49 * 16);
  int rem  = wave - path * (Nj * 49 * 16);
  int n    = rem / (49 * 16);
  rem     -= n * (49 * 16);
  int mt = rem >> 4;
  int ct = rem & 15;
  int lane = threadIdx.x & 31;
  int half = lane >> 4;
  int l    = lane & 15;
  int col  = ct * 16 + l;

  const __bf16* Xrow = X2 + (((size_t)path * Nj + n) * Mrows + (size_t)(mt * 16 + l)) * K2;

  v8f acc[3] = {};
  for (int k0 = 0; k0 < K2; k0 += 32) {
    const __bf16* ap = Xrow + k0 + half * 8;
    bf16x8 alo = *(const bf16x8*)(ap);
    bf16x8 ahi = *(const bf16x8*)(ap + 16);
    bf16x16 a = __builtin_shufflevector(alo, ahi,
        0,1,2,3,4,5,6,7,8,9,10,11,12,13,14,15);
#pragma unroll
    for (int q = 0; q < 3; ++q) {
      int g2 = path * 3 + q;
      const __bf16* bp = W2 + (((size_t)g2 * Nj + n) * Hh + col) * K2 + k0 + half * 8;
      bf16x8 blo = *(const bf16x8*)(bp);
      bf16x8 bhi = *(const bf16x8*)(bp + 16);
      bf16x16 bfrag = __builtin_shufflevector(blo, bhi,
          0,1,2,3,4,5,6,7,8,9,10,11,12,13,14,15);
      acc[q] = __builtin_amdgcn_wmma_f32_16x16x32_bf16(
          false, a, false, bfrag, (short)0, acc[q], false, false);
    }
  }

  // ---- fused global-cell epilogue ----
  float b0 = bg[((size_t)(path * 3 + 0) * Nj + n) * Hh + col];
  float b1 = bg[((size_t)(path * 3 + 1) * Nj + n) * Hh + col];
  float b2 = bg[((size_t)(path * 3 + 2) * Nj + n) * Hh + col];
  const float* cold = path ? c_g_s : c_g_t;

#pragma unroll
  for (int r = 0; r < 8; ++r) {
    int row = mt * 16 + half * 8 + r;
    long long idx = ((long long)row * Nj + n) * Hh + col;
    float f    = sigf (acc[0][r] + b0);
    float cand = tanhf(acc[1][r] + b1);
    float o    = sigf (acc[2][r] + b2);
    float cnew = f * cold[idx] + (1.0f - f) * cand;
    float gnew = o * tanhf(cnew);
    out[(long long)(2 + path * 2) * E + idx] = gnew;  // g_{t|s}_new
    out[(long long)(3 + path * 2) * E + idx] = cnew;  // c_g_{t|s}_new
  }
}

// ---------------------------------------------------------------------------
// Launch
// ---------------------------------------------------------------------------
extern "C" void kernel_launch(void* const* d_in, const int* in_sizes, int n_in,
                              void* d_out, int out_size, void* d_ws, size_t ws_size,
                              hipStream_t stream) {
  const float* h     = (const float*)d_in[0];
  const float* c_h   = (const float*)d_in[1];
  const float* p     = (const float*)d_in[2];
  const float* g_t   = (const float*)d_in[3];
  const float* c_g_t = (const float*)d_in[4];
  const float* g_s   = (const float*)d_in[5];
  const float* c_g_s = (const float*)d_in[6];
  const float* U     = (const float*)d_in[7];
  const float* Wt    = (const float*)d_in[8];
  const float* Ws    = (const float*)d_in[9];
  const float* Zt    = (const float*)d_in[10];
  const float* Zs    = (const float*)d_in[11];
  const float* bgt   = (const float*)d_in[12];
  const float* Wg    = (const float*)d_in[13];
  const float* Zg    = (const float*)d_in[14];
  const float* bgg   = (const float*)d_in[15];
  float* out = (float*)d_out;

  char* ws = (char*)d_ws;
  __bf16* Wb   = (__bf16*)(ws + OFF_WB);
  __bf16* Wb2  = (__bf16*)(ws + OFF_WB2);
  __bf16* Xb   = (__bf16*)(ws + OFF_XB);
  __bf16* Xb2  = (__bf16*)(ws + OFF_XB2);

  const int TPB = 256;
  // --- prep (f32 -> bf16, weight transpose, activation concat) ---
  prep_w1<<<(unsigned)((SZ_WB  + TPB - 1) / TPB), TPB, 0, stream>>>(U, Wt, Ws, Zt, Zs, Wb);
  prep_w2<<<(unsigned)((SZ_WB2 + TPB - 1) / TPB), TPB, 0, stream>>>(Wg, Zg, Wb2);
  prep_x1<<<(unsigned)((SZ_XB  + TPB - 1) / TPB), TPB, 0, stream>>>(p, h, g_t, g_s, Xb);
  prep_x2g<<<(unsigned)((2 * E + TPB - 1) / TPB), TPB, 0, stream>>>(g_t, g_s, Xb2);

  // --- fused pass 1: 24 joints x 49 x 16 tiles, 9 gates/wave, K=1792 ---
  {
    int waves = Nj * 49 * 16;                 // 18816
    int blocks = (waves + (TPB / 32) - 1) / (TPB / 32);
    fused_gate_gemm<<<blocks, TPB, 0, stream>>>(Xb, Wb, bgt, c_h, c_g_t, c_g_s, out, Xb2);
  }

  // --- fused pass 2: 2 paths x 24 joints x 49 x 16 tiles, 3 gates/wave, K=512 ---
  {
    int waves = 2 * Nj * 49 * 16;             // 37632
    int blocks = (waves + (TPB / 32) - 1) / (TPB / 32);
    fused_global_gemm<<<blocks, TPB, 0, stream>>>(Xb2, Wb2, bgg, c_g_t, c_g_s, out);
  }
}
